// transition_decoder_35716948034278
// MI455X (gfx1250) — matrix operand order repaired and checked by reading, not compiled, not run on hardware
//
#include <hip/hip_runtime.h>
#include <hip/hip_bf16.h>

// ---------------------------------------------------------------------------
// N=8, F=512, H=W=96, C=21; conv1 512->256 3x3 @96x96 (+BN+ReLU), bilinear x2,
// conv2 256->128 3x3 @192x192 (+BN+ReLU), conv3 128->21 1x1 + softmax.
// FLOPs are ~99% the two 3x3 convs -> WMMA f16 implicit GEMM; everything else
// is bandwidth-bound glue (~0.7 GB total -> ~30us at 23.3 TB/s).
// ---------------------------------------------------------------------------
#define NB    8
#define FCH   512
#define HH0   96
#define WW0   96
#define HW0   (HH0*WW0)        // 9216
#define NCLS  21
#define HH2   192
#define WW2   192
#define HW2   (HH2*WW2)        // 36864
#define O1    256
#define O2    128

typedef __attribute__((ext_vector_type(16))) _Float16 v16h;
typedef __attribute__((ext_vector_type(8)))  float    v8f;
typedef __attribute__((ext_vector_type(4)))  int      v4i;

union HFrag { v16h v; uint4 q[2]; unsigned u[8]; };

// ---- async global->LDS (guarded; falls back to sync b128 copies) ----------
#if defined(__HIP_DEVICE_COMPILE__) && \
    __has_builtin(__builtin_amdgcn_global_load_async_to_lds_b128) && \
    __has_builtin(__builtin_amdgcn_s_wait_asynccnt)
#define TD_ASYNC 1
#else
#define TD_ASYNC 0
#endif

__device__ __forceinline__ void td_cp16(const _Float16* g, _Float16* l) {
#if TD_ASYNC
    __builtin_amdgcn_global_load_async_to_lds_b128(
        (__attribute__((address_space(1))) v4i*)(void*)const_cast<_Float16*>(g),
        (__attribute__((address_space(3))) v4i*)(void*)l, 0, 0);
#else
    *(uint4*)l = *(const uint4*)g;
#endif
}
__device__ __forceinline__ void td_async_wait() {
#if TD_ASYNC
    __builtin_amdgcn_s_wait_asynccnt(0);
#endif
}

// ---------------------------------------------------------------------------
// weight transform: OIHW f32 -> tap-major [9][COUT][CIN] f16
// ---------------------------------------------------------------------------
__global__ void td_wxform(const float* __restrict__ w, _Float16* __restrict__ wt,
                          int COUT, int CIN) {
    int g = blockIdx.x * 256 + threadIdx.x;
    int total = COUT * CIN * 9;
    if (g >= total) return;
    int i = g % CIN; int r = g / CIN; int o = r % COUT; int tap = r / COUT;
    wt[g] = (_Float16)w[((size_t)o * CIN + i) * 9 + tap];
}

// ---------------------------------------------------------------------------
// argmax over classes -> label map
// ---------------------------------------------------------------------------
__global__ void td_argmax(const float* __restrict__ gp, int* __restrict__ lbl) {
    int g = blockIdx.x * 256 + threadIdx.x;          // N*HW0 threads
    int n = g / HW0, p = g % HW0;
    const float* q = gp + (size_t)n * NCLS * HW0 + p;
    float best = q[0]; int bi = 0;
    #pragma unroll
    for (int c = 1; c < NCLS; ++c) {
        float v = q[(size_t)c * HW0];
        if (v > best) { best = v; bi = c; }
    }
    lbl[g] = bi;
}

// ---------------------------------------------------------------------------
// transition mask: 1 iff clipped 5x5 label window is NOT uniform
// (dilate-erode of one-hot + clamp collapses to exactly this)
// ---------------------------------------------------------------------------
__global__ void td_transition(const int* __restrict__ lbl, float* __restrict__ trans) {
    int g = blockIdx.x * 256 + threadIdx.x;
    int n = g / HW0, p = g % HW0;
    int y = p / WW0, x = p % WW0;
    const int* lp = lbl + n * HW0;
    int c0 = lp[p];
    bool uniform = true;
    #pragma unroll
    for (int dy = -2; dy <= 2; ++dy) {
        int yy = y + dy; if (yy < 0 || yy >= HH0) continue;
        #pragma unroll
        for (int dx = -2; dx <= 2; ++dx) {
            int xx = x + dx; if (xx < 0 || xx >= WW0) continue;
            uniform = uniform && (lp[yy * WW0 + xx] == c0);
        }
    }
    trans[g] = uniform ? 0.f : 1.f;
}

// ---------------------------------------------------------------------------
// prototypes: objs[n,c,f] = sum_p feature*(trans==0 && lbl==c) / (H*W)
// ---------------------------------------------------------------------------
__global__ void __launch_bounds__(256)
td_objs(const float* __restrict__ feature, const int* __restrict__ lbl,
        const float* __restrict__ trans, float* __restrict__ objs) {
    int n = blockIdx.x >> 9;
    int f = blockIdx.x & 511;
    const float* fp = feature + ((size_t)n * FCH + f) * HW0;
    const int*   lp = lbl   + n * HW0;
    const float* tp = trans + n * HW0;
    float acc[NCLS];
    #pragma unroll
    for (int c = 0; c < NCLS; ++c) acc[c] = 0.f;
    for (int p = threadIdx.x; p < HW0; p += 256) {
        if (tp[p] == 0.f) {
            float v = fp[p];
            int   c = lp[p];
            #pragma unroll
            for (int k = 0; k < NCLS; ++k) acc[k] += (c == k) ? v : 0.f;
        }
    }
    __shared__ float red[256];
    for (int c = 0; c < NCLS; ++c) {
        red[threadIdx.x] = acc[c];
        __syncthreads();
        for (int s = 128; s > 0; s >>= 1) {
            if (threadIdx.x < s) red[threadIdx.x] += red[threadIdx.x + s];
            __syncthreads();
        }
        if (threadIdx.x == 0)
            objs[((size_t)n * NCLS + c) * FCH + f] = red[0] * (1.f / (float)HW0);
        __syncthreads();
    }
}

// ---------------------------------------------------------------------------
// cdist + softmax -> dist[n,p,c] = 1 - softmax_c(||tfeat_p - objs_c||)
// ---------------------------------------------------------------------------
__global__ void __launch_bounds__(256)
td_dist(const float* __restrict__ feature, const float* __restrict__ trans,
        const float* __restrict__ objs, float* __restrict__ dist) {
    __shared__ float so[NCLS * FCH];
    __shared__ float so2[NCLS];
    int n = blockIdx.x / (HW0 / 256);
    int p = (blockIdx.x % (HW0 / 256)) * 256 + threadIdx.x;
    const float* ob = objs + (size_t)n * NCLS * FCH;
    for (int i = threadIdx.x; i < NCLS * FCH; i += 256) so[i] = ob[i];
    __syncthreads();
    if (threadIdx.x < NCLS) {
        float s = 0.f;
        for (int f = 0; f < FCH; ++f) { float v = so[threadIdx.x * FCH + f]; s += v * v; }
        so2[threadIdx.x] = s;
    }
    __syncthreads();

    float tpv = trans[n * HW0 + p];
    const float* fp = feature + (size_t)n * FCH * HW0 + p;
    float dot[NCLS]; float x2 = 0.f;
    #pragma unroll
    for (int c = 0; c < NCLS; ++c) dot[c] = 0.f;
    for (int f = 0; f < FCH; ++f) {
        float v = fp[(size_t)f * HW0] * tpv;
        x2 += v * v;
        #pragma unroll
        for (int c = 0; c < NCLS; ++c) dot[c] += v * so[c * FCH + f];
    }
    float cd[NCLS];
    #pragma unroll
    for (int c = 0; c < NCLS; ++c) {
        float d2 = x2 + so2[c] - 2.f * dot[c];
        d2 = d2 > 0.f ? d2 : 0.f;
        cd[c] = sqrtf(d2);
    }
    float mx = cd[0];
    #pragma unroll
    for (int c = 1; c < NCLS; ++c) mx = fmaxf(mx, cd[c]);
    float s = 0.f;
    #pragma unroll
    for (int c = 0; c < NCLS; ++c) { cd[c] = __expf(cd[c] - mx); s += cd[c]; }
    float inv = 1.f / s;
    float* dp = dist + ((size_t)n * HW0 + p) * NCLS;
    #pragma unroll
    for (int c = 0; c < NCLS; ++c) dp[c] = 1.f - cd[c] * inv;
}

// ---------------------------------------------------------------------------
// fusion (NCHW f16): t? 0.1*feat + 0.9*sum_c dist[p,c]*objs[c,f] : feat
// ---------------------------------------------------------------------------
__global__ void __launch_bounds__(256)
td_fusion(const float* __restrict__ feature, const float* __restrict__ trans,
          const float* __restrict__ objs, const float* __restrict__ dist,
          _Float16* __restrict__ fus) {
    int n = blockIdx.x >> 9;
    int f = blockIdx.x & 511;
    float ofc[NCLS];
    const float* ob = objs + (size_t)n * NCLS * FCH + f;
    #pragma unroll
    for (int c = 0; c < NCLS; ++c) ofc[c] = ob[(size_t)c * FCH];
    const float* fp = feature + ((size_t)n * FCH + f) * HW0;
    const float* tp = trans + (size_t)n * HW0;
    const float* dp = dist + (size_t)n * HW0 * NCLS;
    _Float16* op = fus + ((size_t)n * FCH + f) * HW0;
    for (int p = threadIdx.x; p < HW0; p += 256) {
        float t = tp[p];
        float fv = fp[p];
        float val;
        if (t > 0.f) {
            float of = 0.f;
            #pragma unroll
            for (int c = 0; c < NCLS; ++c) of += dp[p * NCLS + c] * ofc[c];
            val = fv * (1.f - t) + fv * t * 0.1f + of * t * 0.9f;
        } else {
            val = fv;
        }
        op[p] = (_Float16)val;
    }
}

// ---------------------------------------------------------------------------
// 32x32 LDS transpose: [C][HW] f16 -> [HW][C] f16 (per image)
// ---------------------------------------------------------------------------
__global__ void __launch_bounds__(256)
td_tr16(const _Float16* __restrict__ in, _Float16* __restrict__ out, int C, int HW) {
    __shared__ _Float16 T[32][33];
    int tilesP = HW >> 5, tilesC = C >> 5;
    int b = blockIdx.x;
    int n = b / (tilesP * tilesC);
    int r = b % (tilesP * tilesC);
    int ct = r / tilesP, pt = r % tilesP;
    const _Float16* ip = in + (size_t)n * C * HW;
    _Float16* op = out + (size_t)n * (size_t)C * HW;
    int cB = ct * 32, pB = pt * 32;
    int pl = threadIdx.x & 31, cl = threadIdx.x >> 5;
    #pragma unroll
    for (int j = 0; j < 4; ++j)
        T[cl + j * 8][pl] = ip[(size_t)(cB + cl + j * 8) * HW + pB + pl];
    __syncthreads();
    int c2 = threadIdx.x & 31, p2 = threadIdx.x >> 5;
    #pragma unroll
    for (int j = 0; j < 4; ++j)
        op[(size_t)(pB + p2 + j * 8) * C + cB + c2] = T[c2][p2 + j * 8];
}

// ---------------------------------------------------------------------------
// Implicit-GEMM 3x3 SAME conv + BN + ReLU, WMMA f32_16x16x32_f16.
// Input NHWC [N,HT,WT,CIN]; weights tap-major [9][COUT][CIN].
// Block tile: 32 x-contiguous pixels x 64 ochans; 8 wave32 waves = 2Mx4N grid
// of 16x16 accumulators. Per CIN-chunk: stage 3-row halo (A) + all 9 taps of
// B once, then 9 unrolled WMMAs. All staging is 16-byte copies (async LDS
// when available); all fragment loads are ds_load_b128.
// ---------------------------------------------------------------------------
template <int CIN, int COUT, int HT, int WT, bool OUT_NHWC>
__global__ void __launch_bounds__(256)
td_conv3x3_wmma(const _Float16* __restrict__ in,
                const _Float16* __restrict__ wgt,
                const float* __restrict__ bng, const float* __restrict__ bnb,
                const float* __restrict__ bnm, const float* __restrict__ bnv,
                _Float16* __restrict__ out) {
    constexpr int TILES_X = WT / 32;
    constexpr int NBLK = COUT / 64;
    constexpr int BLOCKS_PER_IMG = HT * TILES_X * NBLK;
    constexpr int KCN = CIN / 32;

    int bi = blockIdx.x;
    int n  = bi / BLOCKS_PER_IMG;
    int r  = bi % BLOCKS_PER_IMG;
    int nb = r / (HT * TILES_X);
    int r2 = r % (HT * TILES_X);
    int y  = r2 / TILES_X;
    int xt = r2 % TILES_X;
    int xBase = xt * 32;
    int nBase = nb * 64;

    __shared__ _Float16 Ah[3][34][32];    // [row][x(+halo)][cin]  64B x-stride
    __shared__ _Float16 Bs[64][9 * 32];   // [ochan][tap*32 + cin] 576B rows

    const int tid   = threadIdx.x;
    const int lane  = tid & 31;
    const int waveM = (tid >> 5) & 1;
    const int waveN = tid >> 6;
    const int l15   = lane & 15;
    const bool lo   = lane < 16;

    v8f acc = {};

    for (int kc = 0; kc < KCN; ++kc) {
        // ---- stage A halo: 3 rows x 34 x-sites x 4 16B-chunks = 408 chunks
        #pragma unroll
        for (int pass = 0; pass < 2; ++pass) {
            int c = pass * 256 + tid;
            if (c < 408) {
                int row = c / 136, rem = c % 136;
                int xx = rem >> 2, kg = rem & 3;
                int yy = y + row - 1, gx = xBase + xx - 1;
                _Float16* lp = &Ah[row][xx][kg * 8];
                if (yy >= 0 && yy < HT && gx >= 0 && gx < WT) {
                    const _Float16* gp = in +
                        ((size_t)(n * HT + yy) * WT + gx) * CIN + kc * 32 + kg * 8;
                    td_cp16(gp, lp);
                    if (kc + 1 < KCN) __builtin_prefetch(gp + 32, 0, 1);
                } else {
                    uint4 z = {0, 0, 0, 0};
                    *(uint4*)lp = z;
                }
            }
        }
        // ---- stage B: 64 ochan x 9 taps x 4 16B-chunks = 2304 chunks
        #pragma unroll
        for (int j = 0; j < 9; ++j) {
            int c = j * 256 + tid;
            int o = c / 36, rem = c % 36;
            int tap = rem >> 2, kg = rem & 3;
            const _Float16* gp = wgt +
                ((size_t)(tap * COUT + nBase + o)) * CIN + kc * 32 + kg * 8;
            td_cp16(gp, &Bs[o][tap * 32 + kg * 8]);
        }
        td_async_wait();
        __syncthreads();

        // ---- 9 taps, fully unrolled: 4x ds_load_b128 + 1x v_wmma each
        #pragma unroll
        for (int tap = 0; tap < 9; ++tap) {
            const int ky = tap / 3 - 1, kx = tap % 3 - 1;
            int xh = waveM * 16 + l15 + kx + 1;
            HFrag a, b;
            a.q[0] = *(const uint4*)&Ah[ky + 1][xh][lo ? 0 : 8];
            a.q[1] = *(const uint4*)&Ah[ky + 1][xh][lo ? 16 : 24];
            int ncol = waveN * 16 + l15;
            b.q[0] = *(const uint4*)&Bs[ncol][tap * 32 + (lo ? 0 : 8)];
            b.q[1] = *(const uint4*)&Bs[ncol][tap * 32 + (lo ? 16 : 24)];
            acc = __builtin_amdgcn_wmma_f32_16x16x32_f16(false, a.v, false, b.v,
                                                         (short)0, acc, false, false);
        }
        __syncthreads();
    }

    // ---- epilogue: BN + ReLU
    int o = nBase + waveN * 16 + l15;
    float sc = bng[o] * rsqrtf(bnv[o] + 1e-5f);
    float sh = bnb[o] - bnm[o] * sc;
    if (!OUT_NHWC) {
        // NCHW: lane's 8 values are x-contiguous -> one 16B store
        union { _Float16 h[8]; uint4 q; } pk;
        #pragma unroll
        for (int rr = 0; rr < 8; ++rr) {
            float v = acc[rr] * sc + sh;
            v = v > 0.f ? v : 0.f;
            pk.h[rr] = (_Float16)v;
        }
        size_t base = ((size_t)n * COUT + o) * (HT * WT) + (size_t)y * WT
                    + xBase + waveM * 16 + (lo ? 0 : 8);
        *(uint4*)&out[base] = pk.q;
    } else {
        #pragma unroll
        for (int rr = 0; rr < 8; ++rr) {
            int m = lo ? rr : (rr + 8);
            int px = xBase + waveM * 16 + m;
            float v = acc[rr] * sc + sh;
            v = v > 0.f ? v : 0.f;
            out[((size_t)(n * HT + y) * WT + px) * COUT + o] = (_Float16)v;
        }
    }
}

// ---------------------------------------------------------------------------
// bilinear x2 (half-pixel centers), NCHW f16 -> NCHW f16
// (conv1 output is 37.7MB -> L2-resident at 192MB, reads stay cheap)
// ---------------------------------------------------------------------------
__global__ void td_upsample(const _Float16* __restrict__ in, _Float16* __restrict__ out) {
    size_t g = (size_t)blockIdx.x * 256 + threadIdx.x;   // N*O1*HW2 threads
    int ox = (int)(g % WW2); size_t r = g / WW2;
    int oy = (int)(r % HH2); r /= HH2;
    int c  = (int)(r % O1);
    int n  = (int)(r / O1);
    float sy = oy * 0.5f - 0.25f, sx = ox * 0.5f - 0.25f;
    int y0 = (int)floorf(sy), x0 = (int)floorf(sx);
    float fy = sy - (float)y0, fx = sx - (float)x0;
    int y0c = y0 < 0 ? 0 : y0, y1c = (y0 + 1 > HH0 - 1) ? (HH0 - 1) : (y0 + 1);
    int x0c = x0 < 0 ? 0 : x0, x1c = (x0 + 1 > WW0 - 1) ? (WW0 - 1) : (x0 + 1);
    const _Float16* ip = in + ((size_t)n * O1 + c) * HW0;
    float v00 = (float)ip[y0c * WW0 + x0c], v01 = (float)ip[y0c * WW0 + x1c];
    float v10 = (float)ip[y1c * WW0 + x0c], v11 = (float)ip[y1c * WW0 + x1c];
    float v = v00 * (1.f - fy) * (1.f - fx) + v01 * (1.f - fy) * fx
            + v10 * fy * (1.f - fx) + v11 * fy * fx;
    out[g] = (_Float16)v;
}

// ---------------------------------------------------------------------------
// 1x1 conv (128->21) + bias + softmax; input NHWC -> b128 channel loads
// ---------------------------------------------------------------------------
__global__ void __launch_bounds__(256)
td_conv1x1_softmax(const _Float16* __restrict__ in,  // [N, 192, 192, 128] NHWC
                   const float* __restrict__ w3,     // [21, 128]
                   const float* __restrict__ b3,     // [21]
                   float* __restrict__ out) {        // [N, 21, 192, 192]
    __shared__ float ws[NCLS * O2];
    __shared__ float bs[NCLS];
    for (int i = threadIdx.x; i < NCLS * O2; i += 256) ws[i] = w3[i];
    if (threadIdx.x < NCLS) bs[threadIdx.x] = b3[threadIdx.x];
    __syncthreads();
    int g = blockIdx.x * 256 + threadIdx.x;   // N*HW2 threads
    int n = g / HW2, p = g % HW2;
    const uint4* ip4 = (const uint4*)(in + ((size_t)n * HW2 + p) * O2);
    float lg[NCLS];
    #pragma unroll
    for (int c = 0; c < NCLS; ++c) lg[c] = bs[c];
    #pragma unroll 4
    for (int ch = 0; ch < O2 / 8; ++ch) {
        uint4 u = ip4[ch];
        const _Float16* hp = (const _Float16*)&u;
        #pragma unroll
        for (int j = 0; j < 8; ++j) {
            float x = (float)hp[j];
            int i = ch * 8 + j;
            #pragma unroll
            for (int c = 0; c < NCLS; ++c) lg[c] += ws[c * O2 + i] * x;
        }
    }
    float mx = lg[0];
    #pragma unroll
    for (int c = 1; c < NCLS; ++c) mx = fmaxf(mx, lg[c]);
    float s = 0.f;
    #pragma unroll
    for (int c = 0; c < NCLS; ++c) { lg[c] = __expf(lg[c] - mx); s += lg[c]; }
    float inv = 1.f / s;
    float* op = out + (size_t)n * NCLS * HW2 + p;
    #pragma unroll
    for (int c = 0; c < NCLS; ++c) op[(size_t)c * HW2] = lg[c] * inv;
}

// ---------------------------------------------------------------------------
// launch
// ---------------------------------------------------------------------------
extern "C" void kernel_launch(void* const* d_in, const int* in_sizes, int n_in,
                              void* d_out, int out_size, void* d_ws, size_t ws_size,
                              hipStream_t stream) {
    const float* feature = (const float*)d_in[0];
    const float* gpred   = (const float*)d_in[1];
    const float* c1w     = (const float*)d_in[2];
    const float* bn1g    = (const float*)d_in[3];
    const float* bn1b    = (const float*)d_in[4];
    const float* bn1m    = (const float*)d_in[5];
    const float* bn1v    = (const float*)d_in[6];
    const float* c2w     = (const float*)d_in[7];
    const float* bn2g    = (const float*)d_in[8];
    const float* bn2b    = (const float*)d_in[9];
    const float* bn2m    = (const float*)d_in[10];
    const float* bn2v    = (const float*)d_in[11];
    const float* c3w     = (const float*)d_in[12];
    const float* c3b     = (const float*)d_in[13];
    float* outp          = (float*)d_out;
    (void)ws_size; (void)in_sizes; (void)n_in; (void)out_size;

    char* base = (char*)d_ws;
    size_t off = 0;
    auto alloc = [&](size_t bytes) {
        char* p = base + off;
        off += (bytes + 255) & ~(size_t)255;
        return p;
    };
    const size_t szFus = (size_t)NB * FCH * HW0 * 2;   // 75,497,472 (256-mult)
    const size_t szC1  = (size_t)NB * O1 * HW0 * 2;    // 37,748,736
    const size_t szUp  = (size_t)NB * O1 * HW2 * 2;    // 150,994,944 == 2*szFus

    int*      lbl   = (int*)      alloc((size_t)NB * HW0 * 4);
    float*    trans = (float*)    alloc((size_t)NB * HW0 * 4);
    float*    objs  = (float*)    alloc((size_t)NB * NCLS * FCH * 4);
    float*    dist  = (float*)    alloc((size_t)NB * HW0 * NCLS * 4);
    _Float16* w1t   = (_Float16*) alloc((size_t)9 * O1 * FCH * 2);
    _Float16* w2t   = (_Float16*) alloc((size_t)9 * O2 * O1 * 2);
    _Float16* fusN  = (_Float16*) alloc(szFus);        // A (NCHW fusion)
    _Float16* fusT  = (_Float16*) alloc(szFus);        // B (NHWC fusion) = A+szFus
    _Float16* c1out = (_Float16*) alloc(szC1);         // C (NCHW conv1 out)
    _Float16* upN   = (_Float16*) alloc(szUp);         // D (NCHW upsampled)
    _Float16* upT   = fusN;    // alias A+B contiguous (151MB), fusion dead by then
    _Float16* c2out = upN;     // alias D, upN dead once conv2 starts reading upT

    // 1) weight transforms (tap-major f16)
    td_wxform<<<(O1 * FCH * 9 + 255) / 256, 256, 0, stream>>>(c1w, w1t, O1, FCH);
    td_wxform<<<(O2 * O1 * 9 + 255) / 256, 256, 0, stream>>>(c2w, w2t, O2, O1);
    // 2) labels + transition mask
    td_argmax<<<NB * HW0 / 256, 256, 0, stream>>>(gpred, lbl);
    td_transition<<<NB * HW0 / 256, 256, 0, stream>>>(lbl, trans);
    // 3) prototypes, 4) cdist+softmax, 5) fusion (NCHW f16)
    td_objs<<<NB * FCH, 256, 0, stream>>>(feature, lbl, trans, objs);
    td_dist<<<NB * (HW0 / 256), 256, 0, stream>>>(feature, trans, objs, dist);
    td_fusion<<<NB * FCH, 256, 0, stream>>>(feature, trans, objs, dist, fusN);
    // 6) NCHW -> NHWC for conv1
    td_tr16<<<NB * (HW0 / 32) * (FCH / 32), 256, 0, stream>>>(fusN, fusT, FCH, HW0);
    // 7) conv1 (WMMA) -> NCHW
    td_conv3x3_wmma<FCH, O1, HH0, WW0, false>
        <<<NB * HH0 * (WW0 / 32) * (O1 / 64), 256, 0, stream>>>(
            fusT, w1t, bn1g, bn1b, bn1m, bn1v, c1out);
    // 8) bilinear x2 (NCHW), 9) NCHW -> NHWC for conv2
    td_upsample<<<(int)(((size_t)NB * O1 * HW2) / 256), 256, 0, stream>>>(c1out, upN);
    td_tr16<<<NB * (HW2 / 32) * (O1 / 32), 256, 0, stream>>>(upN, upT, O1, HW2);
    // 10) conv2 (WMMA) -> NHWC
    td_conv3x3_wmma<O1, O2, HH2, WW2, true>
        <<<NB * HH2 * (WW2 / 32) * (O2 / 64), 256, 0, stream>>>(
            upT, w2t, bn2g, bn2b, bn2m, bn2v, c2out);
    // 11) 1x1 conv + class softmax -> output (NCHW f32)
    td_conv1x1_softmax<<<NB * HW2 / 256, 256, 0, stream>>>(c2out, c3w, c3b, outp);
}
